// RNN_16801912062326
// MI455X (gfx1250) — matrix operand order repaired and checked by reading, not compiled, not run on hardware
//
#include <hip/hip_runtime.h>
#include <hip/hip_bf16.h>

typedef __attribute__((ext_vector_type(16))) _Float16 v16h;
typedef __attribute__((ext_vector_type(8)))  float    v8f;

#define B_    256
#define T_    512
#define DIN_  128
#define H_    256
#define DOUT_ 64
#define K1_   (DIN_ + H_)
#define LEAK_ 0.01f

#define BT    16          // batch rows per workgroup
#define WIH_S 386         // padded LDS row strides (halves) -> odd DWORD stride, no bank conflicts
#define WHO_S 258
#define H_S   258
#define U_S   130

// A-fragment: 16x32 f16, lane holds row M=lane&15; VGPR0-3 K=half*8+0..7, VGPR4-7 K=16+half*8+0..7
// (8 consecutive halves twice -> compiler emits 2x ds_load_b128)
__device__ __forceinline__ v16h frag_a(const _Float16* base, int stride, int k0, int lane) {
  const _Float16* row = base + (lane & 15) * stride;
  const int hl = (lane >> 4) * 8;
  v16h a;
#pragma unroll
  for (int v = 0; v < 8; ++v) {
    int k = k0 + ((v & 4) << 2) + hl + ((v & 3) << 1);
    a[2 * v]     = row[k];
    a[2 * v + 1] = row[k + 1];
  }
  return a;
}

// B-fragment: 32x16 f16 (K x N), lane holds col N=lane&15; VGPR0-3 K=half*8+0..7, VGPR4-7 K=16+half*8+0..7
// B[k][n] = W[n][k]  (we multiply by W^T), W stored row-major [N][K] in LDS
__device__ __forceinline__ v16h frag_b(const _Float16* base, int stride, int n0, int k0, int lane) {
  const _Float16* row = base + (n0 + (lane & 15)) * stride;
  const int hk = (lane >> 4) * 8;
  v16h b;
#pragma unroll
  for (int v = 0; v < 8; ++v) {
    int k = k0 + ((v & 4) << 2) + hk + ((v & 3) << 1);
    b[2 * v]     = row[k];
    b[2 * v + 1] = row[k + 1];
  }
  return b;
}

__global__ __launch_bounds__(256, 1)
void rnn_scan_kernel(const float* __restrict__ inputs,
                     const float* __restrict__ x0,
                     const float* __restrict__ h0,
                     const float* __restrict__ W_ih,
                     const float* __restrict__ b_ih,
                     const float* __restrict__ W_ho,
                     const float* __restrict__ b_ho,
                     float* __restrict__ xOut,
                     float* __restrict__ hOut) {
  extern __shared__ _Float16 smem[];
  _Float16* sWih = smem;                    // 256 x 386
  _Float16* sWho = sWih + H_ * WIH_S;       // 64 x 258
  _Float16* sHa  = sWho + DOUT_ * WHO_S;    // 16 x 258 (ping)
  _Float16* sHb  = sHa + BT * H_S;          // 16 x 258 (pong)
  _Float16* sU   = sHb + BT * H_S;          // 16 x 130
  float*    sBih = (float*)(sU + BT * U_S); // 256 f32
  float*    sBho = sBih + H_;               // 64 f32

  const int tid  = threadIdx.x;
  const int lane = tid & 31;
  const int wave = tid >> 5;
  const int b0   = blockIdx.x * BT;

  // ---- one-time: weights + biases to LDS (f32 -> f16) ----
  for (int i = tid; i < H_ * K1_; i += 256) {
    int r = i / K1_, c = i - r * K1_;
    sWih[r * WIH_S + c] = (_Float16)W_ih[i];
  }
  for (int i = tid; i < DOUT_ * H_; i += 256) {
    int r = i >> 8, c = i & 255;
    sWho[r * WHO_S + c] = (_Float16)W_ho[i];
  }
  if (tid < H_)    sBih[tid] = b_ih[tid];
  if (tid < DOUT_) sBho[tid] = b_ho[tid];

  // ---- t = 0: emit initial states, seed hidden buffer ----
  for (int i = tid; i < BT * H_; i += 256) {
    int m = i >> 8, c = i & 255;
    float v = h0[(size_t)(b0 + m) * H_ + c];
    sHa[m * H_S + c] = (_Float16)v;
    __builtin_nontemporal_store(v, &hOut[((size_t)(b0 + m) * (T_ + 1)) * H_ + c]);
  }
  for (int i = tid; i < BT * DOUT_; i += 256) {
    int m = i >> 6, c = i & 63;
    __builtin_nontemporal_store(x0[(size_t)(b0 + m) * DOUT_ + c],
                                &xOut[((size_t)(b0 + m) * (T_ + 1)) * DOUT_ + c]);
  }

  // ---- stage u(0) (f32 -> f16) ----
  for (int i = tid; i < BT * DIN_; i += 256) {
    int m = i >> 7, c = i & 127;
    sU[m * U_S + c] = (_Float16)inputs[((size_t)(b0 + m) * T_) * DIN_ + c];
  }

  for (int t = 0; t < T_; ++t) {
    __syncthreads();  // barrier A: u(t) staged, h(t) complete, everyone past prior reads
    _Float16* hR = (t & 1) ? sHb : sHa;   // read
    _Float16* hW = (t & 1) ? sHa : sHb;   // write

    // ---- GEMM1: pre = [u_t, h] @ W_ih^T ----
    // two N-tiles per wave (nt = wave, wave+8) fused in one K-loop: A loaded once, two WMMAs
    {
      v8f acc0 = {}, acc1 = {};
#pragma unroll
      for (int kk = 0; kk < K1_ / 32; ++kk) {
        v16h a  = (kk < DIN_ / 32) ? frag_a(sU, U_S, kk * 32, lane)
                                   : frag_a(hR, H_S, (kk - DIN_ / 32) * 32, lane);
        v16h bL = frag_b(sWih, WIH_S, wave * 16,       kk * 32, lane);
        v16h bH = frag_b(sWih, WIH_S, (wave + 8) * 16, kk * 32, lane);
        acc0 = __builtin_amdgcn_wmma_f32_16x16x32_f16(false, a, false, bL, (short)0, acc0, false, false);
        acc1 = __builtin_amdgcn_wmma_f32_16x16x32_f16(false, a, false, bH, (short)0, acc1, false, false);
      }
      const int col0 = wave * 16 + (lane & 15);
      const int col1 = col0 + 128;
      const float bi0 = sBih[col0];
      const float bi1 = sBih[col1];
#pragma unroll
      for (int v = 0; v < 8; ++v) {
        int m = (lane >> 4) * 8 + v;           // C/D layout: row = half*8 + vgpr
        float pre0 = acc0[v] + bi0;
        float pre1 = acc1[v] + bi1;
        float hn0  = pre0 >= 0.0f ? pre0 : LEAK_ * pre0;
        float hn1  = pre1 >= 0.0f ? pre1 : LEAK_ * pre1;
        hW[m * H_S + col0] = (_Float16)hn0;
        hW[m * H_S + col1] = (_Float16)hn1;
        size_t gbase = ((size_t)(b0 + m) * (T_ + 1) + (t + 1)) * H_;
        __builtin_nontemporal_store(hn0, &hOut[gbase + col0]);
        __builtin_nontemporal_store(hn1, &hOut[gbase + col1]);
      }
    }
    __syncthreads();  // barrier B: h_new fully in hW, all sU reads done

    if (wave < 4) {
      // ---- GEMM2: x_t = h_new @ W_ho^T ; 4 tiles on waves 0..3 ----
      const int nt = wave;
      v8f acc = {};
#pragma unroll
      for (int kk = 0; kk < H_ / 32; ++kk) {
        v16h a = frag_a(hW, H_S, kk * 32, lane);
        v16h b = frag_b(sWho, WHO_S, nt * 16, kk * 32, lane);
        acc = __builtin_amdgcn_wmma_f32_16x16x32_f16(false, a, false, b, (short)0, acc, false, false);
      }
      const int col  = nt * 16 + (lane & 15);
      const float bo = sBho[col];
#pragma unroll
      for (int v = 0; v < 8; ++v) {
        int m = (lane >> 4) * 8 + v;
        __builtin_nontemporal_store(acc[v] + bo,
            &xOut[((size_t)(b0 + m) * (T_ + 1) + (t + 1)) * DOUT_ + col]);
      }
    } else if (t + 1 < T_) {
      // ---- waves 4..7: stage u(t+1) under GEMM2 (sU free after barrier B) ----
      for (int i = tid - 128; i < BT * DIN_; i += 128) {
        int m = i >> 7, c = i & 127;
        const float* p = &inputs[((size_t)(b0 + m) * T_ + (t + 1)) * DIN_ + c];
        sU[m * U_S + c] = (_Float16)(*p);
        if (t + 2 < T_) __builtin_prefetch(p + DIN_, 0, 0);
      }
    }
    // loop-top barrier A orders staging/GEMM2 against GEMM1(t+1)
  }
}

extern "C" void kernel_launch(void* const* d_in, const int* in_sizes, int n_in,
                              void* d_out, int out_size, void* d_ws, size_t ws_size,
                              hipStream_t stream) {
  const float* inputs = (const float*)d_in[0];
  const float* x0     = (const float*)d_in[1];
  const float* h0     = (const float*)d_in[2];
  const float* W_ih   = (const float*)d_in[3];
  const float* b_ih   = (const float*)d_in[4];
  const float* W_ho   = (const float*)d_in[5];
  const float* b_ho   = (const float*)d_in[6];

  float* xOut = (float*)d_out;                              // [B, T+1, DOUT]
  float* hOut = xOut + (size_t)B_ * (T_ + 1) * DOUT_;       // [B, T+1, H]

  const size_t shmem =
      (size_t)(H_ * WIH_S + DOUT_ * WHO_S + 2 * BT * H_S + BT * U_S) * sizeof(short)
      + (size_t)(H_ + DOUT_) * sizeof(float);               // ~253 KB of the 320 KB WGP LDS

  dim3 grid(B_ / BT), block(256);
  rnn_scan_kernel<<<grid, block, shmem, stream>>>(inputs, x0, h0, W_ih, b_ih,
                                                  W_ho, b_ho, xOut, hOut);
}
